// DS_Block_57269093924947
// MI455X (gfx1250) — matrix operand order repaired and checked by reading, not compiled, not run on hardware
//
#include <hip/hip_runtime.h>
#include <hip/hip_bf16.h>
#include <math.h>

// ---------------------------------------------------------------------------
// Problem constants (from reference)
// ---------------------------------------------------------------------------
#define B_  4
#define C_  128
#define N_  4000
#define H_  4
#define K_  9
#define EPS_IN 1e-3f
#define EPS_BN 1e-5f
#define EPS_LN 1e-5f

typedef __attribute__((ext_vector_type(16))) _Float16 v16h;
typedef __attribute__((ext_vector_type(8)))  _Float16 v8h;
typedef __attribute__((ext_vector_type(8)))  float    v8f;

// ---------------------------------------------------------------------------
// WMMA fragment helpers (CDNA5 wave32, V_WMMA_F32_16X16X32_F16)
//
// A (16x32 f16, row-major source A[M][ldk]):
//   lane = kh*16+m : elems 0..15 = K[k0 + kh*16 .. +15]
// B (32x16 f16, column-major source BT[col][ldk]):
//   lane = kh*16+n : elems 0..15 = K[k0 + kh*16 .. +15] of column c0+n
// C/D (16x16 f32): vgpr i, lane = g*16+n -> row = 8*g + i, col = n
// ---------------------------------------------------------------------------
__device__ __forceinline__ v16h frag_a(const _Float16* A, int ldk, int r0, int k0) {
    int lane = threadIdx.x & 31;
    int m = lane & 15, kh = lane >> 4;
    const _Float16* base = A + (size_t)(r0 + m) * ldk + k0;
    v8h lo = *(const v8h*)(base + kh * 16);
    v8h hi = *(const v8h*)(base + kh * 16 + 8);
    v16h f;
#pragma unroll
    for (int i = 0; i < 8; ++i) { f[i] = lo[i]; f[8 + i] = hi[i]; }
    return f;
}

__device__ __forceinline__ v16h frag_b(const _Float16* BT, int ldk, int c0, int k0) {
    int lane = threadIdx.x & 31;
    int n = lane & 15, kh = lane >> 4;
    const _Float16* base = BT + (size_t)(c0 + n) * ldk + k0 + kh * 16;
    v8h lo = *(const v8h*)(base);
    v8h hi = *(const v8h*)(base + 8);
    v16h f;
#pragma unroll
    for (int i = 0; i < 8; ++i) { f[i] = lo[i]; f[8 + i] = hi[i]; }
    return f;
}

__device__ __forceinline__ v8f wmma_f16(v16h a, v16h b, v8f c) {
    return __builtin_amdgcn_wmma_f32_16x16x32_f16(false, a, false, b, (short)0, c,
                                                  false, false);
}

// ---------------------------------------------------------------------------
// f32 -> f16 flat conversion (weights)
// ---------------------------------------------------------------------------
__global__ void f32_to_f16_kernel(const float* __restrict__ src,
                                  _Float16* __restrict__ dst, int n) {
    int i = blockIdx.x * blockDim.x + threadIdx.x;
    if (i < n) dst[i] = (_Float16)src[i];
}

// ---------------------------------------------------------------------------
// LayerNorm over channel dim; writes f32 [b][c][n] and f16 transposed [b][n][c]
// one wave per point, 8 points per 256-thread block
// ---------------------------------------------------------------------------
__global__ __launch_bounds__(256)
void ln_kernel(const float* __restrict__ x, const float* __restrict__ g,
               const float* __restrict__ bb, float* __restrict__ xs,
               _Float16* __restrict__ xsT) {
    int wave = threadIdx.x >> 5, lane = threadIdx.x & 31;
    int pt = blockIdx.x * 8 + wave;
    if (pt >= B_ * N_) return;
    int b = pt / N_, n = pt - b * N_;
    const float* xp = x + (size_t)b * C_ * N_ + n;
    float v[4], s = 0.f, s2 = 0.f;
#pragma unroll
    for (int i = 0; i < 4; ++i) {
        float t = xp[(size_t)(lane * 4 + i) * N_];
        v[i] = t; s += t; s2 += t * t;
    }
#pragma unroll
    for (int off = 16; off; off >>= 1) {
        s  += __shfl_xor(s,  off, 32);
        s2 += __shfl_xor(s2, off, 32);
    }
    float m = s / (float)C_;
    float inv = rsqrtf(s2 / (float)C_ - m * m + EPS_LN);
#pragma unroll
    for (int i = 0; i < 4; ++i) {
        int c = lane * 4 + i;
        float o = (v[i] - m) * inv * g[c] + bb[c];
        xs[(size_t)b * C_ * N_ + (size_t)c * N_ + n] = o;
        xsT[((size_t)b * N_ + n) * C_ + c] = (_Float16)o;
    }
}

// ---------------------------------------------------------------------------
// Generic WMMA GEMM:  Out[b][o][n] = sum_k A[o][k] * BT[b][n][k]  (M = 128)
// mode 0: Out = acc + bias
// mode 1: Out = relu((acc+bias) * g[o]/sqrt(1+eps_bn) + be[o])     (dgcnn w2)
// mode 2: Out = acc + bias + resA + resB                           (proj+resid)
// grid (ceil(250/4), 8, B), block 128 (4 waves, one 16x16 tile each)
// ---------------------------------------------------------------------------
__global__ __launch_bounds__(128)
void gemm128_kernel(const _Float16* __restrict__ A, const _Float16* __restrict__ BTall,
                    int Kd, const float* __restrict__ bias,
                    const float* __restrict__ g, const float* __restrict__ be,
                    const float* __restrict__ resA, const float* __restrict__ resB,
                    float* __restrict__ Out, int mode) {
    int wave = threadIdx.x >> 5;
    int c0 = (blockIdx.x * 4 + wave) * 16;
    if (c0 >= N_) return;                 // wave-uniform: EXEC stays all-ones
    int r0 = blockIdx.y * 16;
    int b  = blockIdx.z;
    const _Float16* BT = BTall + (size_t)b * N_ * Kd;

    v8f acc = (v8f){0, 0, 0, 0, 0, 0, 0, 0};
    for (int k0 = 0; k0 < Kd; k0 += 32) {
        if (k0 + 32 < Kd)
            __builtin_prefetch(BT + (size_t)c0 * Kd + k0 + 32, 0, 1);
        v16h af = frag_a(A, Kd, r0, k0);
        v16h bf = frag_b(BT, Kd, c0, k0);
        acc = wmma_f16(af, bf, acc);
    }

    int lane = threadIdx.x & 31;
    int nn = lane & 15, mb = (lane >> 4) * 8;
    float inv_bn = rsqrtf(1.f + EPS_BN);
#pragma unroll
    for (int i = 0; i < 8; ++i) {
        int o = r0 + mb + i;
        size_t e = (size_t)b * C_ * N_ + (size_t)o * N_ + c0 + nn;
        float t = acc[i] + bias[o];
        if (mode == 1) t = fmaxf(t * (g[o] * inv_bn) + be[o], 0.f);
        else if (mode == 2) t = t + resA[e] + resB[e];
        Out[e] = t;
    }
}

// ---------------------------------------------------------------------------
// Instance-norm statistics per (b, c) over N
// ---------------------------------------------------------------------------
__global__ __launch_bounds__(256)
void inorm_stats_kernel(const float* __restrict__ Y, float* __restrict__ mean,
                        float* __restrict__ inv) {
    int bc = blockIdx.x;
    const float* y = Y + (size_t)bc * N_;
    __shared__ float ss[256], ss2[256];
    float s = 0.f, s2 = 0.f;
    for (int n = threadIdx.x; n < N_; n += 256) { float t = y[n]; s += t; s2 += t * t; }
    ss[threadIdx.x] = s; ss2[threadIdx.x] = s2;
    __syncthreads();
    for (int o = 128; o; o >>= 1) {
        if (threadIdx.x < o) { ss[threadIdx.x] += ss[threadIdx.x + o];
                               ss2[threadIdx.x] += ss2[threadIdx.x + o]; }
        __syncthreads();
    }
    if (threadIdx.x == 0) {
        float m = ss[0] / (float)N_;
        mean[bc] = m;
        inv[bc] = rsqrtf(ss2[0] / (float)N_ - m * m + EPS_IN);
    }
}

// ---------------------------------------------------------------------------
// Instance-norm + BN apply.
// mode 0: o = relu((y-m)*iv*gs + be); write OutF32 (opt) and OutT f16 [b][n][c]
// mode 1: t = (y-m)*iv*gs + be;  OutFinal = res2 + relu(t + res1)
// ---------------------------------------------------------------------------
__global__ __launch_bounds__(256)
void inorm_apply_kernel(const float* __restrict__ Y, const float* __restrict__ mean,
                        const float* __restrict__ inv, const float* __restrict__ g,
                        const float* __restrict__ be, float* __restrict__ OutF32,
                        _Float16* __restrict__ OutT, const float* __restrict__ res1,
                        const float* __restrict__ res2, float* __restrict__ OutFinal,
                        int mode) {
    size_t i = (size_t)blockIdx.x * 256 + threadIdx.x;
    size_t total = (size_t)B_ * C_ * N_;
    if (i >= total) return;
    int n = (int)(i % N_);
    int bc = (int)(i / N_);
    int c = bc % C_;
    int b = bc / C_;
    float gs = g[c] * rsqrtf(1.f + EPS_BN);
    float t = (Y[i] - mean[bc]) * inv[bc] * gs + be[c];
    if (mode == 0) {
        t = fmaxf(t, 0.f);
        if (OutF32) OutF32[i] = t;
        if (OutT) OutT[((size_t)b * N_ + n) * C_ + c] = (_Float16)t;
    } else {
        OutFinal[i] = res2[i] + fmaxf(t + res1[i], 0.f);
    }
}

// ---------------------------------------------------------------------------
// Squared norms per point from f16 [b][n][c]
// ---------------------------------------------------------------------------
__global__ __launch_bounds__(256)
void sq_kernel(const _Float16* __restrict__ xT, float* __restrict__ sq) {
    int i = blockIdx.x * 256 + threadIdx.x;
    if (i >= B_ * N_) return;
    const _Float16* p = xT + (size_t)i * C_;
    float s = 0.f;
    for (int c = 0; c < C_; ++c) { float t = (float)p[c]; s += t * t; }
    sq[i] = s;
}

// ---------------------------------------------------------------------------
// kNN: Gram matrix tiles via WMMA, fused running top-9 per row in LDS.
// pd[n][m] = 2*inner - sq[n] - sq[m]; keep K_ largest (self included -> pd=0).
// grid (N/16, B), block 128 (4 waves -> 64 columns per sweep)
// ---------------------------------------------------------------------------
__global__ __launch_bounds__(128)
void knn_kernel(const _Float16* __restrict__ xTall, const float* __restrict__ sqall,
                int* __restrict__ idx) {
    int b = blockIdx.y;
    int r0 = blockIdx.x * 16;
    const _Float16* X = xTall + (size_t)b * N_ * C_;
    const float* sq = sqall + (size_t)b * N_;

    __shared__ float pd[16][64];
    __shared__ float topv[16][K_];
    __shared__ int   topi[16][K_];
    int tid = threadIdx.x, wave = tid >> 5, lane = tid & 31;

    if (tid < 16)
        for (int j = 0; j < K_; ++j) { topv[tid][j] = -3.4e38f; topi[tid][j] = 0; }
    __syncthreads();

    // A fragments for this 16-row strip are invariant across the sweep
    v16h afr[4];
#pragma unroll
    for (int kc = 0; kc < 4; ++kc) afr[kc] = frag_a(X, C_, r0, kc * 32);

    for (int m0 = 0; m0 < N_; m0 += 64) {
        int c0 = m0 + wave * 16;
        if (c0 < N_) {                       // wave-uniform guard
            v8f acc = (v8f){0, 0, 0, 0, 0, 0, 0, 0};
#pragma unroll
            for (int kc = 0; kc < 4; ++kc)
                acc = wmma_f16(afr[kc], frag_b(X, C_, c0, kc * 32), acc);
            int nn = lane & 15, mb = (lane >> 4) * 8;
            float sm = sq[c0 + nn];
#pragma unroll
            for (int i = 0; i < 8; ++i) {
                int rr = mb + i;
                pd[rr][wave * 16 + nn] = 2.f * acc[i] - sq[r0 + rr] - sm;
            }
        }
        __syncthreads();
        if (tid < 16) {
            int lim = (N_ - m0 < 64) ? (N_ - m0) : 64;
            for (int jj = 0; jj < lim; ++jj) {
                float v = pd[tid][jj];
                if (v > topv[tid][K_ - 1]) {
                    int p = K_ - 1;
                    while (p > 0 && topv[tid][p - 1] < v) {
                        topv[tid][p] = topv[tid][p - 1];
                        topi[tid][p] = topi[tid][p - 1];
                        --p;
                    }
                    topv[tid][p] = v;
                    topi[tid][p] = m0 + jj;
                }
            }
        }
        __syncthreads();
    }
    if (tid < 16)
        for (int j = 0; j < K_; ++j)
            idx[((size_t)b * N_ + r0 + tid) * K_ + j] = topi[tid][j];
}

// ---------------------------------------------------------------------------
// DGCNN stage 1: graph-feature gather + w1 edge-conv GEMM (K=768) via WMMA.
// Per block: 16 points -> 48 (n,j) columns (3 tiles), 8 row tiles of 16.
// Stages 16 center + 144 neighbor vectors (f16) in LDS, builds B fragments
// on the fly: k = c*3+t, feat = c<128 ? ctr[c] : ctr[c-128]-nb[j*3+t][c-128].
// Writes yT f16 [b][n][o*3+j]  (K-layout for the w2 GEMM).
// grid (N/16, B), block 96 (3 waves, one column-tile each)
// ---------------------------------------------------------------------------
__global__ __launch_bounds__(96)
void dgcnn1_kernel(const _Float16* __restrict__ xTall, const int* __restrict__ idx,
                   const _Float16* __restrict__ W1h, const float* __restrict__ b1,
                   const float* __restrict__ g1, const float* __restrict__ be1,
                   _Float16* __restrict__ yT) {
    __shared__ _Float16 nbuf[16][10][C_];   // 40 KB: slot 0 = center, 1..9 = nbrs
    int b = blockIdx.y;
    int n0 = blockIdx.x * 16;
    const _Float16* X = xTall + (size_t)b * N_ * C_;

    for (int v = threadIdx.x; v < 160; v += 96) {
        int nn = v / 10, slot = v - nn * 10;
        int src = (slot == 0) ? (n0 + nn)
                              : idx[((size_t)b * N_ + n0 + nn) * K_ + (slot - 1)];
        const v8h* s = (const v8h*)(X + (size_t)src * C_);
        v8h* d = (v8h*)(&nbuf[nn][slot][0]);
#pragma unroll
        for (int q = 0; q < 16; ++q) d[q] = s[q];
    }
    __syncthreads();

    int wave = threadIdx.x >> 5, lane = threadIdx.x & 31;
    int cl = wave * 16 + (lane & 15);       // column 0..47
    int nn = cl / 3, j = cl - nn * 3;
    int kh = lane >> 4;

    v8f acc[8];
#pragma unroll
    for (int r = 0; r < 8; ++r) acc[r] = (v8f){0, 0, 0, 0, 0, 0, 0, 0};

    for (int k0 = 0; k0 < 768; k0 += 32) {
        v16h bf;
#pragma unroll
        for (int i = 0; i < 16; ++i) {
            int k = k0 + kh * 16 + i;
            int c = k / 3, t = k - c * 3;
            _Float16 val;
            if (c < C_) val = nbuf[nn][0][c];
            else        val = (_Float16)(nbuf[nn][0][c - C_] -
                                         nbuf[nn][1 + j * 3 + t][c - C_]);
            bf[i] = val;
        }
#pragma unroll
        for (int rt = 0; rt < 8; ++rt) {
            v16h af = frag_a(W1h, 768, rt * 16, k0);
            acc[rt] = wmma_f16(af, bf, acc[rt]);
        }
    }

    float inv_bn = rsqrtf(1.f + EPS_BN);
    int mb = (lane >> 4) * 8;
#pragma unroll
    for (int rt = 0; rt < 8; ++rt)
#pragma unroll
        for (int i = 0; i < 8; ++i) {
            int o = rt * 16 + mb + i;
            float t = acc[rt][i] + b1[o];
            t = fmaxf(t * (g1[o] * inv_bn) + be1[o], 0.f);
            yT[(((size_t)b * N_ + n0 + nn) * C_ + o) * 3 + j] = (_Float16)t;
        }
}

// ---------------------------------------------------------------------------
// L2 scale per (b, c) over N: scale = 1/max(||row||, 1e-12)
// ---------------------------------------------------------------------------
__global__ __launch_bounds__(256)
void l2scale_kernel(const float* __restrict__ Y, float* __restrict__ scale) {
    int bc = blockIdx.x;
    __shared__ float sh[256];
    float s = 0.f;
    for (int n = threadIdx.x; n < N_; n += 256) {
        float t = Y[(size_t)bc * N_ + n];
        s += t * t;
    }
    sh[threadIdx.x] = s;
    __syncthreads();
    for (int o = 128; o; o >>= 1) {
        if (threadIdx.x < o) sh[threadIdx.x] += sh[threadIdx.x + o];
        __syncthreads();
    }
    if (threadIdx.x == 0) scale[bc] = 1.f / fmaxf(sqrtf(sh[0]), 1e-12f);
}

// ---------------------------------------------------------------------------
// Per-head channel attention: S = (qh kh^T)*T, softmax rows, out = S vh.
// S is only 32x32 per (b,h) -> one block per head, LDS-tiled over n.
// Writes f16 transposed out [b][n][c] for the proj GEMM.
// ---------------------------------------------------------------------------
__global__ __launch_bounds__(256)
void attn_kernel(const float* __restrict__ qd, const float* __restrict__ kd,
                 const float* __restrict__ vd, const float* __restrict__ qs,
                 const float* __restrict__ ks, const float* __restrict__ temp,
                 _Float16* __restrict__ outT) {
    int bh = blockIdx.x;
    int b = bh / H_, h = bh - b * H_;
    const float* Q = qd + ((size_t)b * C_ + h * 32) * N_;
    const float* Kp = kd + ((size_t)b * C_ + h * 32) * N_;
    const float* V = vd + ((size_t)b * C_ + h * 32) * N_;
    __shared__ float S[32][32];
    __shared__ float qt[32][64], kt[32][64];
    int tid = threadIdx.x;

    float accp[4] = {0.f, 0.f, 0.f, 0.f};
    for (int n0 = 0; n0 < N_; n0 += 64) {
        for (int e = tid; e < 32 * 64; e += 256) {
            int cc = e >> 6, nn = e & 63;
            int n = n0 + nn;
            float qv = (n < N_) ? Q[(size_t)cc * N_ + n] : 0.f;
            float kv = (n < N_) ? Kp[(size_t)cc * N_ + n] : 0.f;
            qt[cc][nn] = qv;
            kt[cc][nn] = kv;
        }
        __syncthreads();
#pragma unroll
        for (int p = 0; p < 4; ++p) {
            int pair = tid * 4 + p;
            int ci = pair >> 5, di = pair & 31;
            float s = 0.f;
            for (int nn = 0; nn < 64; ++nn) s += qt[ci][nn] * kt[di][nn];
            accp[p] += s;
        }
        __syncthreads();
    }
    float T = temp[h];
#pragma unroll
    for (int p = 0; p < 4; ++p) {
        int pair = tid * 4 + p;
        int ci = pair >> 5, di = pair & 31;
        S[ci][di] = accp[p] * qs[(size_t)b * C_ + h * 32 + ci] *
                    ks[(size_t)b * C_ + h * 32 + di] * T;
    }
    __syncthreads();
    if (tid < 32) {
        float mx = -3.4e38f;
        for (int d = 0; d < 32; ++d) mx = fmaxf(mx, S[tid][d]);
        float sm = 0.f;
        for (int d = 0; d < 32; ++d) { float e = __expf(S[tid][d] - mx); S[tid][d] = e; sm += e; }
        float r = 1.f / sm;
        for (int d = 0; d < 32; ++d) S[tid][d] *= r;
    }
    __syncthreads();
    for (int e = tid; e < 32 * 4096; e += 256) {
        int ci = e >> 12, n = e & 4095;
        if (n >= N_) continue;
        float s = 0.f;
#pragma unroll
        for (int d = 0; d < 32; ++d) s += S[ci][d] * V[(size_t)d * N_ + n];
        int c = h * 32 + ci;
        outT[((size_t)b * N_ + n) * C_ + c] = (_Float16)s;
    }
}

// ---------------------------------------------------------------------------
// Host orchestration
// ---------------------------------------------------------------------------
// setup_inputs flattening (dict order, nested dicts in insertion order):
//  0 x | 1 ln1_g 2 ln1_b 3 ln2_g 4 ln2_b 5 temperature
//  6..9   qf{w,b,g,beta}  10..13 kf{..}  14..17 vf{..}
//  18..25 gq{w1,b1,g1,be1,w2,b2,g2,be2}  26..33 gk{..}  34..41 gv{..}
//  42 proj_w 43 proj_b | 44..47 f1{w,b,g,be} | 48..51 f2{w,b,g,be}
extern "C" void kernel_launch(void* const* d_in, const int* in_sizes, int n_in,
                              void* d_out, int out_size, void* d_ws, size_t ws_size,
                              hipStream_t stream) {
    (void)in_sizes; (void)n_in; (void)out_size; (void)ws_size;
    const float* x      = (const float*)d_in[0];
    const float* ln1_g  = (const float*)d_in[1];
    const float* ln1_b  = (const float*)d_in[2];
    const float* ln2_g  = (const float*)d_in[3];
    const float* ln2_b  = (const float*)d_in[4];
    const float* temp   = (const float*)d_in[5];
    const float* fw[3]  = {(const float*)d_in[6],  (const float*)d_in[10], (const float*)d_in[14]};
    const float* fb[3]  = {(const float*)d_in[7],  (const float*)d_in[11], (const float*)d_in[15]};
    const float* fg[3]  = {(const float*)d_in[8],  (const float*)d_in[12], (const float*)d_in[16]};
    const float* fbe[3] = {(const float*)d_in[9],  (const float*)d_in[13], (const float*)d_in[17]};
    const float* gw1[3] = {(const float*)d_in[18], (const float*)d_in[26], (const float*)d_in[34]};
    const float* gb1[3] = {(const float*)d_in[19], (const float*)d_in[27], (const float*)d_in[35]};
    const float* gg1[3] = {(const float*)d_in[20], (const float*)d_in[28], (const float*)d_in[36]};
    const float* gbe1[3]= {(const float*)d_in[21], (const float*)d_in[29], (const float*)d_in[37]};
    const float* gw2[3] = {(const float*)d_in[22], (const float*)d_in[30], (const float*)d_in[38]};
    const float* gb2[3] = {(const float*)d_in[23], (const float*)d_in[31], (const float*)d_in[39]};
    const float* gg2[3] = {(const float*)d_in[24], (const float*)d_in[32], (const float*)d_in[40]};
    const float* gbe2[3]= {(const float*)d_in[25], (const float*)d_in[33], (const float*)d_in[41]};
    const float* projw  = (const float*)d_in[42];
    const float* projb  = (const float*)d_in[43];
    const float* f1w = (const float*)d_in[44];
    const float* f1b = (const float*)d_in[45];
    const float* f1g = (const float*)d_in[46];
    const float* f1be= (const float*)d_in[47];
    const float* f2w = (const float*)d_in[48];
    const float* f2b = (const float*)d_in[49];
    const float* f2g = (const float*)d_in[50];
    const float* f2be= (const float*)d_in[51];

    // workspace carve-up (256B aligned)
    char* base = (char*)d_ws;
    size_t cur = 0;
    auto alloc = [&](size_t bytes) -> void* {
        void* p = base + cur;
        cur += (bytes + 255) & ~(size_t)255;
        return p;
    };
    const size_t ACT = (size_t)B_ * C_ * N_;          // 2,048,000 elems

    _Float16* fwh[3];   for (int i=0;i<3;++i) fwh[i]  = (_Float16*)alloc(C_*C_*2);
    _Float16* gw1h[3];  for (int i=0;i<3;++i) gw1h[i] = (_Float16*)alloc(C_*2*C_*3*2);
    _Float16* gw2h[3];  for (int i=0;i<3;++i) gw2h[i] = (_Float16*)alloc(C_*C_*3*2);
    _Float16* projwh = (_Float16*)alloc(C_*C_*2);
    _Float16* f1wh   = (_Float16*)alloc(C_*C_*2);
    _Float16* f2wh   = (_Float16*)alloc(C_*C_*2);

    float*    xs1   = (float*)alloc(ACT * 4);
    _Float16* xs1T  = (_Float16*)alloc(ACT * 2);
    float*    tmp   = (float*)alloc(ACT * 4);
    _Float16* filtT[3]; for (int i=0;i<3;++i) filtT[i] = (_Float16*)alloc(ACT * 2);
    float*    sqb   = (float*)alloc((size_t)B_ * N_ * 4);
    int*      idxb  = (int*)alloc((size_t)B_ * N_ * K_ * 4);
    _Float16* yT    = (_Float16*)alloc((size_t)B_ * N_ * C_ * 3 * 2);
    float*    dg[3]; for (int i=0;i<3;++i) dg[i] = (float*)alloc(ACT * 4);
    float*    qs    = (float*)alloc((size_t)B_ * C_ * 4);
    float*    ks    = (float*)alloc((size_t)B_ * C_ * 4);
    _Float16* attT  = (_Float16*)alloc(ACT * 2);
    float*    x1    = (float*)alloc(ACT * 4);
    float*    xs2   = (float*)alloc(ACT * 4);
    _Float16* xs2T  = (_Float16*)alloc(ACT * 2);
    _Float16* h1T   = (_Float16*)alloc(ACT * 2);
    float*    meanb = (float*)alloc((size_t)B_ * C_ * 4);
    float*    invb  = (float*)alloc((size_t)B_ * C_ * 4);

    auto cvt = [&](const float* s, _Float16* d, int n) {
        f32_to_f16_kernel<<<(n + 255) / 256, 256, 0, stream>>>(s, d, n);
    };
    for (int i = 0; i < 3; ++i) {
        cvt(fw[i],  fwh[i],  C_ * C_);
        cvt(gw1[i], gw1h[i], C_ * 2 * C_ * 3);
        cvt(gw2[i], gw2h[i], C_ * C_ * 3);
    }
    cvt(projw, projwh, C_ * C_);
    cvt(f1w, f1wh, C_ * C_);
    cvt(f2w, f2wh, C_ * C_);

    dim3 gemm_grid((N_ / 16 + 3) / 4, C_ / 16, B_);
    int apply_blocks = (int)((ACT + 255) / 256);

    // xs1 = LN1(x)
    ln_kernel<<<(B_ * N_) / 8, 256, 0, stream>>>(x, ln1_g, ln1_b, xs1, xs1T);

    // q/k/v branches: filter -> knn -> edge-conv -> w2 conv
    for (int i = 0; i < 3; ++i) {
        gemm128_kernel<<<gemm_grid, 128, 0, stream>>>(
            fwh[i], xs1T, C_, fb[i], nullptr, nullptr, nullptr, nullptr, tmp, 0);
        inorm_stats_kernel<<<B_ * C_, 256, 0, stream>>>(tmp, meanb, invb);
        inorm_apply_kernel<<<apply_blocks, 256, 0, stream>>>(
            tmp, meanb, invb, fg[i], fbe[i], nullptr, filtT[i],
            nullptr, nullptr, nullptr, 0);
        sq_kernel<<<(B_ * N_ + 255) / 256, 256, 0, stream>>>(filtT[i], sqb);
        knn_kernel<<<dim3(N_ / 16, B_), 128, 0, stream>>>(filtT[i], sqb, idxb);
        dgcnn1_kernel<<<dim3(N_ / 16, B_), 96, 0, stream>>>(
            filtT[i], idxb, gw1h[i], gb1[i], gg1[i], gbe1[i], yT);
        gemm128_kernel<<<gemm_grid, 128, 0, stream>>>(
            gw2h[i], yT, C_ * 3, gb2[i], gg2[i], gbe2[i], nullptr, nullptr,
            dg[i], 1);
    }

    // attention (channel-wise, per head)
    l2scale_kernel<<<B_ * C_, 256, 0, stream>>>(dg[0], qs);
    l2scale_kernel<<<B_ * C_, 256, 0, stream>>>(dg[1], ks);
    attn_kernel<<<B_ * H_, 256, 0, stream>>>(dg[0], dg[1], dg[2], qs, ks, temp, attT);

    // x1 = x + proj(attn_out) + proj_b + xs1
    gemm128_kernel<<<gemm_grid, 128, 0, stream>>>(
        projwh, attT, C_, projb, nullptr, nullptr, xs1, x, x1, 2);

    // FFN
    ln_kernel<<<(B_ * N_) / 8, 256, 0, stream>>>(x1, ln2_g, ln2_b, xs2, xs2T);

    gemm128_kernel<<<gemm_grid, 128, 0, stream>>>(
        f1wh, xs2T, C_, f1b, nullptr, nullptr, nullptr, nullptr, tmp, 0);
    inorm_stats_kernel<<<B_ * C_, 256, 0, stream>>>(tmp, meanb, invb);
    inorm_apply_kernel<<<apply_blocks, 256, 0, stream>>>(
        tmp, meanb, invb, f1g, f1be, nullptr, h1T, nullptr, nullptr, nullptr, 0);

    gemm128_kernel<<<gemm_grid, 128, 0, stream>>>(
        f2wh, h1T, C_, f2b, nullptr, nullptr, nullptr, nullptr, tmp, 0);
    inorm_stats_kernel<<<B_ * C_, 256, 0, stream>>>(tmp, meanb, invb);
    // out = x1 + relu(bn(inorm(tmp)) + xs2)
    inorm_apply_kernel<<<apply_blocks, 256, 0, stream>>>(
        tmp, meanb, invb, f2g, f2be, nullptr, nullptr, xs2, x1, (float*)d_out, 1);
}
